// GAT_75436805587126
// MI455X (gfx1250) — matrix operand order repaired and checked by reading, not compiled, not run on hardware
//
#include <hip/hip_runtime.h>
#include <hip/hip_bf16.h>

typedef __attribute__((ext_vector_type(16))) _Float16 v16h;
typedef __attribute__((ext_vector_type(8)))  float    v8f;

constexpr int N_   = 50000;
constexpr int E_   = 800000;
constexpr int FIN_ = 128;
constexpr int H_   = 3;
constexpr int C_   = 64;
constexpr int HC_  = H_ * C_;     // 192
constexpr int XW_  = FIN_ + 2;    // 130 (col0 = mask, cols1..128 = features)
constexpr float NEG_SLOPE_ = 0.2f;

// ---- ordered-uint encoding for float atomic max (monotone map) ----
__device__ __forceinline__ unsigned f2ord(float f) {
    unsigned u = __float_as_uint(f);
    return (u & 0x80000000u) ? ~u : (u | 0x80000000u);
}
__device__ __forceinline__ float ord2f(unsigned u) {
    return (u & 0x80000000u) ? __uint_as_float(u ^ 0x80000000u)
                             : __uint_as_float(~u);
}
// f2ord(-inf) = 0x007FFFFF -> init sentinel

// ================= init =================
__global__ void k_init(unsigned* __restrict__ mx, float* __restrict__ denom,
                       float* __restrict__ out, int n3, int nOut) {
    int idx = blockIdx.x * blockDim.x + threadIdx.x;
    if (idx < nOut) out[idx] = 0.0f;
    if (idx < n3) { mx[idx] = 0x007FFFFFu; denom[idx] = 0.0f; }
}

// ====== kernel 1: h = relu((x@Wcom+bcom)*(1-m) + (x@Wtoll+btoll)*m) =====
// block = 128 threads (4 waves); block handles 16 node rows; wave w -> cols [16w,16w+16)
__global__ __launch_bounds__(128) void k_node_embed(
    const float* __restrict__ x,
    const float* __restrict__ Wcom,  const float* __restrict__ bcom,
    const float* __restrict__ Wtoll, const float* __restrict__ btoll,
    float* __restrict__ h)
{
    const int wave = threadIdx.x >> 5;
    const int lane = threadIdx.x & 31;
    const int rowBase = blockIdx.x * 16;
    const int col  = (wave << 4) + (lane & 15);
    const int arow = rowBase + (lane & 15);
    const int akb  = (lane >> 4) * 16;   // A frag K sub-offset (0 or 16)
    const int bkb  = (lane >> 4) * 16;   // B frag K sub-offset (0 or 16)

    v8f accC = {}; v8f accT = {};
    for (int kb = 0; kb < FIN_; kb += 32) {
        // A fragment: 16-bit A 16x32 layout (x rows only 4B-aligned -> scalar loads)
        v16h a = {};
        const float* xr = x + (size_t)arow * XW_ + 1 + kb + akb;
        #pragma unroll
        for (int j = 0; j < 8; ++j) {
            a[j]     = (_Float16)xr[j];
            a[8 + j] = (_Float16)xr[8 + j];
        }
        // B fragments: col = lane&15, K rows kb+bkb .. +15 (coalesced across lanes)
        v16h bc = {}, bt = {};
        const float* wc = Wcom  + (size_t)(kb + bkb) * C_ + col;
        const float* wt = Wtoll + (size_t)(kb + bkb) * C_ + col;
        #pragma unroll
        for (int j = 0; j < 16; ++j) {
            bc[j] = (_Float16)wc[(size_t)j * C_];
            bt[j] = (_Float16)wt[(size_t)j * C_];
        }
        accC = __builtin_amdgcn_wmma_f32_16x16x32_f16(false, a, false, bc, (short)0, accC, false, false);
        accT = __builtin_amdgcn_wmma_f32_16x16x32_f16(false, a, false, bt, (short)0, accT, false, false);
    }
    const float bC = bcom[col], bT = btoll[col];
    #pragma unroll
    for (int v = 0; v < 8; ++v) {
        const int row  = rowBase + v + ((lane >> 4) << 3);   // C/D layout
        const float mv = x[(size_t)row * XW_];
        const float val = (accC[v] + bC) * (1.0f - mv) + (accT[v] + bT) * mv;
        h[(size_t)row * C_ + col] = fmaxf(val, 0.0f);
    }
}

// ====== kernel 2: hs = h@Wsrc, hd = h@Wdst (both 64 -> 192) ======
// block = 256 threads (8 waves); 16 rows per block; 24 col-tiles (12 hs + 12 hd), 3 per wave
__global__ __launch_bounds__(256) void k_headproj(
    const float* __restrict__ h,
    const float* __restrict__ Wsrc, const float* __restrict__ Wdst,
    float* __restrict__ hs, float* __restrict__ hdm)
{
    const int wave = threadIdx.x >> 5;
    const int lane = threadIdx.x & 31;
    const int rowBase = blockIdx.x * 16;
    const int arow = rowBase + (lane & 15);
    const int akb  = (lane >> 4) * 16;
    const int bkb  = (lane >> 4) * 16;

    // preload both A fragments (K = 0..31 and 32..63); h rows are 256B aligned,
    // akb*4 in {0,64}B -> every 8-float run is 16B aligned: use float4 loads.
    v16h a0 = {}, a1 = {};
    {
        const float4* hr = (const float4*)(h + (size_t)arow * C_ + akb);
        // offsets in float4 units: runs at floats {0,8,32,40} -> f4 {0,2,8,10}
        float4 r0a = hr[0], r0b = hr[1];     // K akb+0..7
        float4 r1a = hr[2], r1b = hr[3];     // K akb+8..15
        float4 r2a = hr[8], r2b = hr[9];     // K akb+32..39
        float4 r3a = hr[10], r3b = hr[11];   // K akb+40..47
        a0[0]=(_Float16)r0a.x; a0[1]=(_Float16)r0a.y; a0[2]=(_Float16)r0a.z; a0[3]=(_Float16)r0a.w;
        a0[4]=(_Float16)r0b.x; a0[5]=(_Float16)r0b.y; a0[6]=(_Float16)r0b.z; a0[7]=(_Float16)r0b.w;
        a0[8]=(_Float16)r1a.x; a0[9]=(_Float16)r1a.y; a0[10]=(_Float16)r1a.z; a0[11]=(_Float16)r1a.w;
        a0[12]=(_Float16)r1b.x; a0[13]=(_Float16)r1b.y; a0[14]=(_Float16)r1b.z; a0[15]=(_Float16)r1b.w;
        a1[0]=(_Float16)r2a.x; a1[1]=(_Float16)r2a.y; a1[2]=(_Float16)r2a.z; a1[3]=(_Float16)r2a.w;
        a1[4]=(_Float16)r2b.x; a1[5]=(_Float16)r2b.y; a1[6]=(_Float16)r2b.z; a1[7]=(_Float16)r2b.w;
        a1[8]=(_Float16)r3a.x; a1[9]=(_Float16)r3a.y; a1[10]=(_Float16)r3a.z; a1[11]=(_Float16)r3a.w;
        a1[12]=(_Float16)r3b.x; a1[13]=(_Float16)r3b.y; a1[14]=(_Float16)r3b.z; a1[15]=(_Float16)r3b.w;
    }
    for (int t = wave * 3; t < wave * 3 + 3; ++t) {
        const int isDst = (t >= 12);
        const int c0 = (isDst ? (t - 12) : t) << 4;
        const float* W = isDst ? Wdst : Wsrc;
        const int col = c0 + (lane & 15);
        v8f acc = {};
        v16h b = {};
        #pragma unroll
        for (int j = 0; j < 16; ++j) b[j] = (_Float16)W[(size_t)(bkb + j) * HC_ + col];
        acc = __builtin_amdgcn_wmma_f32_16x16x32_f16(false, a0, false, b, (short)0, acc, false, false);
        #pragma unroll
        for (int j = 0; j < 16; ++j) b[j] = (_Float16)W[(size_t)(32 + bkb + j) * HC_ + col];
        acc = __builtin_amdgcn_wmma_f32_16x16x32_f16(false, a1, false, b, (short)0, acc, false, false);
        float* out = isDst ? hdm : hs;
        #pragma unroll
        for (int v = 0; v < 8; ++v) {
            const int row = rowBase + v + ((lane >> 4) << 3);
            out[(size_t)row * HC_ + col] = acc[v];
        }
    }
}

// ====== kernel 3: per-(node,head) attention dots (float4 vectorized) ======
__global__ void k_att(const float* __restrict__ hs, const float* __restrict__ hdm,
                      const float* __restrict__ attS, const float* __restrict__ attD,
                      float* __restrict__ a_s, float* __restrict__ a_d, int n)
{
    int idx = blockIdx.x * blockDim.x + threadIdx.x;
    if (idx >= n * H_) return;
    const int node = idx / H_, head = idx % H_;
    const float4* ps = (const float4*)(hs  + (size_t)node * HC_ + head * C_);
    const float4* pd = (const float4*)(hdm + (size_t)node * HC_ + head * C_);
    const float4* as = (const float4*)(attS + head * C_);
    const float4* ad = (const float4*)(attD + head * C_);
    float s1 = 0.0f, s2 = 0.0f;
    #pragma unroll
    for (int c = 0; c < C_ / 4; ++c) {
        float4 v1 = ps[c], w1 = as[c];
        float4 v2 = pd[c], w2 = ad[c];
        s1 += v1.x * w1.x + v1.y * w1.y + v1.z * w1.z + v1.w * w1.w;
        s2 += v2.x * w2.x + v2.y * w2.y + v2.z * w2.z + v2.w * w2.w;
    }
    a_s[idx] = s1; a_d[idx] = s2;
}

// ====== kernel 4: edge logits + segment max (ordered-uint atomicMax) ======
__global__ void k_edge_max(const int* __restrict__ ei,
                           const float* __restrict__ a_s, const float* __restrict__ a_d,
                           float* __restrict__ ebuf, unsigned* __restrict__ mx, int e)
{
    int idx = blockIdx.x * blockDim.x + threadIdx.x;
    if (idx >= e) return;
    const int src = ei[idx], dst = ei[e + idx];
    #pragma unroll
    for (int hh = 0; hh < H_; ++hh) {
        float v = a_s[src * H_ + hh] + a_d[dst * H_ + hh];
        v = (v > 0.0f) ? v : NEG_SLOPE_ * v;
        ebuf[(size_t)idx * H_ + hh] = v;
        atomicMax(&mx[dst * H_ + hh], f2ord(v));
    }
}

// ====== kernel 5: exp(e - mx) + segment sum ======
__global__ void k_edge_exp(const int* __restrict__ ei, const unsigned* __restrict__ mx,
                           float* __restrict__ ebuf, float* __restrict__ denom, int e)
{
    int idx = blockIdx.x * blockDim.x + threadIdx.x;
    if (idx >= e) return;
    const int dst = ei[e + idx];
    #pragma unroll
    for (int hh = 0; hh < H_; ++hh) {
        float m = ord2f(mx[dst * H_ + hh]);
        if (!isfinite(m)) m = 0.0f;            // matches reference where(isfinite)
        const float ev = expf(ebuf[(size_t)idx * H_ + hh] - m);
        ebuf[(size_t)idx * H_ + hh] = ev;
        atomicAdd(&denom[dst * H_ + hh], ev);
    }
}

// ====== kernel 6: message aggregation (dominant stage) ======
// One 192-thread block (6 wave32s) per edge: no div/mod, uniform src/dst per
// block, fully coalesced 192-float gather from hs[src] and atomic scatter to
// out[dst].
__global__ __launch_bounds__(192) void k_edge_aggr(
    const int* __restrict__ ei,
    const float* __restrict__ ebuf, const float* __restrict__ denom,
    const float* __restrict__ hs, float* __restrict__ out, int e)
{
    const int edge = blockIdx.x;
    const int elem = threadIdx.x;          // 0..191
    const int head = elem >> 6;            // elem / 64
    const int src = ei[edge], dst = ei[e + edge];
    const float alpha = ebuf[(size_t)edge * H_ + head] /
                        (denom[dst * H_ + head] + 1e-16f);
    atomicAdd(&out[(size_t)dst * HC_ + elem], alpha * hs[(size_t)src * HC_ + elem]);
}

// ====== kernel 7: y = ((out+bias)@Wlin + blin) * mask ; wave32 reduction ======
__global__ __launch_bounds__(256) void k_final(
    const float* __restrict__ out, const float* __restrict__ biasC,
    const float* __restrict__ Wlin, const float* __restrict__ blin,
    const float* __restrict__ x, float* __restrict__ y, int n)
{
    const int node = blockIdx.x * 8 + (threadIdx.x >> 5);
    const int lane = threadIdx.x & 31;
    if (node >= n) return;
    float acc = 0.0f;
    #pragma unroll
    for (int j = lane; j < HC_; j += 32)
        acc += (out[(size_t)node * HC_ + j] + biasC[j]) * Wlin[j];
    #pragma unroll
    for (int off = 16; off > 0; off >>= 1)
        acc += __shfl_down(acc, off, 32);
    if (lane == 0)
        y[node] = (acc + blin[0]) * x[(size_t)node * XW_];
}

extern "C" void kernel_launch(void* const* d_in, const int* in_sizes, int n_in,
                              void* d_out, int out_size, void* d_ws, size_t ws_size,
                              hipStream_t stream) {
    (void)in_sizes; (void)n_in; (void)out_size; (void)ws_size;
    const float* x     = (const float*)d_in[0];
    const int*   ei    = (const int*)  d_in[1];
    const float* Wcom  = (const float*)d_in[2];
    const float* bcom  = (const float*)d_in[3];
    const float* Wtoll = (const float*)d_in[4];
    const float* btoll = (const float*)d_in[5];
    const float* Wsrc  = (const float*)d_in[6];
    const float* Wdst  = (const float*)d_in[7];
    const float* attS  = (const float*)d_in[8];
    const float* attD  = (const float*)d_in[9];
    const float* biasC = (const float*)d_in[10];
    const float* Wlin  = (const float*)d_in[11];
    const float* blin  = (const float*)d_in[12];
    float* y = (float*)d_out;

    char* ws = (char*)d_ws;
    size_t off = 0;
    auto take = [&](size_t bytes) -> char* {
        char* p = ws + off;
        off = (off + bytes + 255) & ~(size_t)255;
        return p;
    };
    float*    h     = (float*)   take((size_t)N_ * C_  * 4);
    float*    hs    = (float*)   take((size_t)N_ * HC_ * 4);
    float*    hdm   = (float*)   take((size_t)N_ * HC_ * 4);
    float*    a_s   = (float*)   take((size_t)N_ * H_  * 4);
    float*    a_d   = (float*)   take((size_t)N_ * H_  * 4);
    unsigned* mx    = (unsigned*)take((size_t)N_ * H_  * 4);
    float*    denom = (float*)   take((size_t)N_ * H_  * 4);
    float*    ebuf  = (float*)   take((size_t)E_ * H_  * 4);
    float*    outb  = (float*)   take((size_t)N_ * HC_ * 4);

    // init accumulators / sentinels
    {
        int nOut = N_ * HC_, n3 = N_ * H_;
        int g = (nOut + 255) / 256;
        k_init<<<g, 256, 0, stream>>>(mx, denom, outb, n3, nOut);
    }
    // node embedding (WMMA)
    k_node_embed<<<N_ / 16, 128, 0, stream>>>(x, Wcom, bcom, Wtoll, btoll, h);
    // head projections (WMMA)
    k_headproj<<<N_ / 16, 256, 0, stream>>>(h, Wsrc, Wdst, hs, hdm);
    // attention dots
    k_att<<<(N_ * H_ + 255) / 256, 256, 0, stream>>>(hs, hdm, attS, attD, a_s, a_d, N_);
    // edge softmax pipeline
    k_edge_max<<<(E_ + 255) / 256, 256, 0, stream>>>(ei, a_s, a_d, ebuf, mx, E_);
    k_edge_exp<<<(E_ + 255) / 256, 256, 0, stream>>>(ei, mx, ebuf, denom, E_);
    // message aggregation: one 192-thread block per edge
    k_edge_aggr<<<E_, 192, 0, stream>>>(ei, ebuf, denom, hs, outb, E_);
    // finalize
    k_final<<<(N_ + 7) / 8, 256, 0, stream>>>(outb, biasC, Wlin, blin, x, y, N_);
}
